// CFSDPNet_89902255440348
// MI455X (gfx1250) — matrix operand order repaired and checked by reading, not compiled, not run on hardware
//
#include <hip/hip_runtime.h>
#include <hip/hip_bf16.h>

#define NPTS 8192
#define DIM  64
#define NBINS 4096
#define D2_RANGE 1024.0f   // histogram d^2 in [0, 32^2]

typedef __attribute__((ext_vector_type(16))) _Float16 v16h;
typedef __attribute__((ext_vector_type(8)))  _Float16 v8h;
typedef __attribute__((ext_vector_type(8)))  float    v8f;

// ---- WMMA fragment loads (per CDNA5 ISA 7.12.2 layouts, wave32) ----
// A (16x32 f16): lane L holds row M=L%16; group g=L/16 selects K runs
//   elements 0..7  -> K = c*32 + g*8 + e
//   elements 8..15 -> K = c*32 + 16 + g*8 + (e-8)
__device__ __forceinline__ v16h load_a_chunk(const _Float16* dh, int row, int c, int g) {
    const _Float16* p = dh + (size_t)row * DIM + c * 32 + g * 8;
    union { v16h v; v8h h[2]; } u;
    u.h[0] = *(const v8h*)(p);
    u.h[1] = *(const v8h*)(p + 16);
    return u.v;
}
// B (32x16 f16): lane L holds col N=L%16 (i.e. data row j0+N); group g=L/16:
//   elements 0..7 -> K = c*32 + g*8 + e; elements 8..15 -> K = c*32 + 16 + g*8 + (e-8)
__device__ __forceinline__ v16h load_b_chunk(const _Float16* dh, int row, int c, int g) {
    const _Float16* p = dh + (size_t)row * DIM + c * 32 + g * 8;
    union { v16h v; v8h h[2]; } u;
    u.h[0] = *(const v8h*)(p);
    u.h[1] = *(const v8h*)(p + 16);
    return u.v;
}
// 16x16 tile of data@data^T for rows [i0,i0+16) x cols [j0,j0+16), K=64.
__device__ __forceinline__ v8f tile_dot(v16h a0, v16h a1, const _Float16* dh, int jrow, int g) {
    v16h b0 = load_b_chunk(dh, jrow, 0, g);
    v16h b1 = load_b_chunk(dh, jrow, 1, g);
    v8f c = {};
    c = __builtin_amdgcn_wmma_f32_16x16x32_f16(false, a0, false, b0, (short)0, c, false, false);
    c = __builtin_amdgcn_wmma_f32_16x16x32_f16(false, a1, false, b1, (short)0, c, false, false);
    return c;
}

// ---- Kernel 1: fp32 -> fp16 copy + squared norms ----
__global__ void prep_kernel(const float* __restrict__ data, _Float16* __restrict__ dh,
                            float* __restrict__ sq) {
    int i = blockIdx.x * blockDim.x + threadIdx.x;
    if (i >= NPTS) return;
    const float* row = data + (size_t)i * DIM;
    _Float16* drow = dh + (size_t)i * DIM;
    float s = 0.f;
    #pragma unroll
    for (int k = 0; k < DIM; ++k) { float v = row[k]; s += v * v; drow[k] = (_Float16)v; }
    sq[i] = s;
}

// ---- Kernel 2: d^2 histogram (for dc percentile; sqrt is monotonic so ranks match) ----
// grid 512 blocks (one 16-row i-tile each), 8 waves sweep all 512 j-tiles.
__global__ void hist_kernel(const _Float16* __restrict__ dh, const float* __restrict__ sq,
                            unsigned int* __restrict__ hist) {
    __shared__ unsigned int lhist[NBINS];
    __shared__ float sqi[16];
    const int t = threadIdx.x;
    const int i0 = blockIdx.x * 16;
    for (int b = t; b < NBINS; b += 256) lhist[b] = 0u;
    if (t < 16) sqi[t] = sq[i0 + t];
    __syncthreads();

    const int wave = t >> 5, lane = t & 31;
    const int g = lane >> 4, n = lane & 15;
    v16h a0 = load_a_chunk(dh, i0 + n, 0, g);
    v16h a1 = load_a_chunk(dh, i0 + n, 1, g);
    float msq[8];
    #pragma unroll
    for (int r = 0; r < 8; ++r) msq[r] = sqi[r + 8 * g];

    const float scale = (float)NBINS / D2_RANGE;
    for (int jt = wave; jt < NPTS / 16; jt += 8) {
        const int j = jt * 16 + n;
        v8f c = tile_dot(a0, a1, dh, j, g);
        const float sqj = sq[j];
        #pragma unroll
        for (int r = 0; r < 8; ++r) {
            float d2 = fmaxf(msq[r] + sqj - 2.0f * c[r], 0.0f);
            int b = (int)(d2 * scale);
            b = b < NBINS - 1 ? b : NBINS - 1;
            atomicAdd(&lhist[b], 1u);
        }
    }
    __syncthreads();
    for (int b = t; b < NBINS; b += 256)
        if (lhist[b]) atomicAdd(&hist[b], lhist[b]);
}

// ---- Kernel 3: select 2nd percentile from d^2 histogram, dc = sqrt(quantile) ----
__global__ void dc_kernel(const unsigned int* __restrict__ hist, float* __restrict__ dcbuf) {
    if (threadIdx.x != 0 || blockIdx.x != 0) return;
    const double target = 0.02 * ((double)NPTS * (double)NPTS - 1.0);
    double cum = 0.0;
    int b = 0;
    for (; b < NBINS - 1; ++b) {
        double c = (double)hist[b];
        if (cum + c > target) break;
        cum += c;
    }
    double cnt = (double)hist[b];
    double frac = (cnt > 0.0) ? (target - cum) / cnt : 0.5;
    const double width = (double)D2_RANGE / (double)NBINS;
    double d2q = ((double)b + frac) * width;
    float dc = (float)sqrt(d2q);
    const float minw = 0.5f * (float)width;
    if (!(dc > 0.0f)) dc = minw;
    dcbuf[0] = dc;
    dcbuf[1] = 1.0f / (dc * dc);
}

// ---- Kernel 4: rho_i = sum_j exp(-d2/dc^2), deterministic reduction ----
__global__ void rho_kernel(const _Float16* __restrict__ dh, const float* __restrict__ sq,
                           const float* __restrict__ dcbuf, float* __restrict__ rho) {
    __shared__ float sqi[16];
    __shared__ float part[8][16];
    const int t = threadIdx.x;
    const int i0 = blockIdx.x * 16;
    if (t < 16) sqi[t] = sq[i0 + t];
    __syncthreads();

    const float inv_dc2 = dcbuf[1];
    const int wave = t >> 5, lane = t & 31;
    const int g = lane >> 4, n = lane & 15;
    v16h a0 = load_a_chunk(dh, i0 + n, 0, g);
    v16h a1 = load_a_chunk(dh, i0 + n, 1, g);
    float msq[8], acc[8];
    #pragma unroll
    for (int r = 0; r < 8; ++r) { msq[r] = sqi[r + 8 * g]; acc[r] = 0.f; }

    for (int jt = wave; jt < NPTS / 16; jt += 8) {
        const int j = jt * 16 + n;
        v8f c = tile_dot(a0, a1, dh, j, g);
        const float sqj = sq[j];
        #pragma unroll
        for (int r = 0; r < 8; ++r) {
            float d2 = fmaxf(msq[r] + sqj - 2.0f * c[r], 0.0f);
            acc[r] += __expf(-d2 * inv_dc2);
        }
    }
    // butterfly over the 16 lanes holding the same rows (fixed order -> deterministic)
    #pragma unroll
    for (int off = 8; off >= 1; off >>= 1)
        #pragma unroll
        for (int r = 0; r < 8; ++r) acc[r] += __shfl_xor(acc[r], off, 32);
    if (n == 0)
        #pragma unroll
        for (int r = 0; r < 8; ++r) part[wave][r + 8 * g] = acc[r];
    __syncthreads();
    if (t < 16) {
        float s = 0.f;
        #pragma unroll
        for (int w = 0; w < 8; ++w) s += part[w][t];
        rho[i0 + t] = s;
    }
}

// ---- Kernel 5: delta_i / nhd_i (min dist to higher-rho point; argmin smallest j) ----
__global__ void delta_kernel(const _Float16* __restrict__ dh, const float* __restrict__ sq,
                             const float* __restrict__ rho, float* __restrict__ delta,
                             int* __restrict__ nhd) {
    __shared__ float sqi[16], rhoi[16];
    __shared__ unsigned long long minkey_l[16];
    __shared__ unsigned int maxd_l[16];
    const int t = threadIdx.x;
    const int i0 = blockIdx.x * 16;
    if (t < 16) {
        sqi[t] = sq[i0 + t]; rhoi[t] = rho[i0 + t];
        minkey_l[t] = ~0ull;  maxd_l[t] = 0u;
    }
    __syncthreads();

    const int wave = t >> 5, lane = t & 31;
    const int g = lane >> 4, n = lane & 15;
    v16h a0 = load_a_chunk(dh, i0 + n, 0, g);
    v16h a1 = load_a_chunk(dh, i0 + n, 1, g);
    float msq[8], mrho[8], mx[8];
    unsigned long long mk[8];
    #pragma unroll
    for (int r = 0; r < 8; ++r) {
        msq[r] = sqi[r + 8 * g]; mrho[r] = rhoi[r + 8 * g];
        mx[r] = 0.f; mk[r] = ~0ull;
    }

    for (int jt = wave; jt < NPTS / 16; jt += 8) {
        const int j = jt * 16 + n;
        v8f c = tile_dot(a0, a1, dh, j, g);
        const float sqj = sq[j];
        const float rhoj = rho[j];
        #pragma unroll
        for (int r = 0; r < 8; ++r) {
            float d2 = fmaxf(msq[r] + sqj - 2.0f * c[r], 0.0f);
            mx[r] = fmaxf(mx[r], d2);
            if (rhoj > mrho[r]) {
                unsigned long long key =
                    ((unsigned long long)__float_as_uint(d2) << 32) | (unsigned int)j;
                mk[r] = key < mk[r] ? key : mk[r];
            }
        }
    }
    #pragma unroll
    for (int r = 0; r < 8; ++r) {
        int m = r + 8 * g;
        atomicMin(&minkey_l[m], mk[r]);
        atomicMax(&maxd_l[m], __float_as_uint(mx[r]));
    }
    __syncthreads();
    if (t < 16) {
        unsigned long long k = minkey_l[t];
        int i = i0 + t;
        if (k == ~0ull) {  // no higher-density point
            delta[i] = sqrtf(__uint_as_float(maxd_l[t]));
            nhd[i] = i;
        } else {
            delta[i] = sqrtf(__uint_as_float((unsigned int)(k >> 32)));
            nhd[i] = (int)(k & 0xffffffffu);
        }
    }
}

// ---- Kernel 6: centers + prefix-scan ranks + pointer-jumping label propagation ----
__global__ void __launch_bounds__(1024)
labels_kernel(const float* __restrict__ rho, const float* __restrict__ delta,
              const int* __restrict__ nhd, const float* __restrict__ rho_t_p,
              const float* __restrict__ delta_t_p, int* __restrict__ out) {
    __shared__ int lab[NPTS];   // 32 KB
    __shared__ int par[NPTS];   // 32 KB
    __shared__ int scan[1024];
    const float rt = rho_t_p[0], dt = delta_t_p[0];
    const int t = threadIdx.x;

    int isC[8];
    int cnt = 0;
    #pragma unroll
    for (int k = 0; k < 8; ++k) {
        int i = t * 8 + k;
        isC[k] = (rho[i] > rt) && (delta[i] > dt);
        cnt += isC[k];
    }
    scan[t] = cnt;
    __syncthreads();
    for (int off = 1; off < 1024; off <<= 1) {
        int v = scan[t];
        int u = (t >= off) ? scan[t - off] : 0;
        __syncthreads();
        scan[t] = v + u;
        __syncthreads();
    }
    int base = (t > 0) ? scan[t - 1] : 0;
    #pragma unroll
    for (int k = 0; k < 8; ++k) {
        int i = t * 8 + k;
        lab[i] = isC[k] ? base : -1;
        base += isC[k];
        int p = nhd[i];
        par[i] = (isC[k] || p == i) ? i : p;  // centers & density-maxima are roots
    }
    __syncthreads();
    // pointer jumping: 2^14 > 8192 covers any chain
    for (int it = 0; it < 14; ++it) {
        int nl[8], np[8];
        #pragma unroll
        for (int k = 0; k < 8; ++k) {
            int i = t * 8 + k;
            int l = lab[i], p = par[i];
            nl[k] = (l < 0) ? lab[p] : l;
            np[k] = par[p];
        }
        __syncthreads();
        #pragma unroll
        for (int k = 0; k < 8; ++k) { int i = t * 8 + k; lab[i] = nl[k]; par[i] = np[k]; }
        __syncthreads();
    }
    #pragma unroll
    for (int k = 0; k < 8; ++k) { int i = t * 8 + k; out[i] = lab[i]; }
}

extern "C" void kernel_launch(void* const* d_in, const int* in_sizes, int n_in,
                              void* d_out, int out_size, void* d_ws, size_t ws_size,
                              hipStream_t stream) {
    const float* data    = (const float*)d_in[0];
    const float* rho_t   = (const float*)d_in[1];
    const float* delta_t = (const float*)d_in[2];
    int* out = (int*)d_out;  // labels are int32

    char* ws = (char*)d_ws;
    _Float16*     dh    = (_Float16*)(ws);                       // 1,048,576 B
    float*        sq    = (float*)(ws + 1048576);                //    32,768 B
    float*        rho   = (float*)(ws + 1048576 + 32768);        //    32,768 B
    float*        delta = (float*)(ws + 1048576 + 65536);        //    32,768 B
    int*          nhd   = (int*)(ws + 1048576 + 98304);          //    32,768 B
    unsigned int* hist  = (unsigned int*)(ws + 1048576 + 131072);//    16,384 B
    float*        dcbuf = (float*)(ws + 1048576 + 147456);       //        16 B
    (void)in_sizes; (void)n_in; (void)out_size; (void)ws_size;

    hipMemsetAsync(hist, 0, NBINS * sizeof(unsigned int), stream);
    prep_kernel<<<(NPTS + 255) / 256, 256, 0, stream>>>(data, dh, sq);
    hist_kernel<<<NPTS / 16, 256, 0, stream>>>(dh, sq, hist);
    dc_kernel<<<1, 64, 0, stream>>>(hist, dcbuf);
    rho_kernel<<<NPTS / 16, 256, 0, stream>>>(dh, sq, dcbuf, rho);
    delta_kernel<<<NPTS / 16, 256, 0, stream>>>(dh, sq, rho, delta, nhd);
    labels_kernel<<<1, 1024, 0, stream>>>(rho, delta, nhd, rho_t, delta_t, out);
}
